// AttentionDecoder_12326556140281
// MI455X (gfx1250) — matrix operand order repaired and checked by reading, not compiled, not run on hardware
//
#include <hip/hip_runtime.h>
#include <hip/hip_bf16.h>

typedef __attribute__((ext_vector_type(16))) __bf16 v16bf;
typedef __attribute__((ext_vector_type(8)))  __bf16 v8bf;
typedef __attribute__((ext_vector_type(8)))  float  v8f;

#define BB 32
#define TT 256
#define LL 784
#define HH 512
#define EE 256
#define VV 8000
#define KCAT 1280   /* E + H (ctx_out) + H (h) */
#define G4H  2048   /* 4*H */

__device__ __forceinline__ float bf2f(unsigned short u) {
  unsigned int x = ((unsigned int)u) << 16;
  float f; __builtin_memcpy(&f, &x, 4); return f;
}
__device__ __forceinline__ float bflo(unsigned int u) {
  unsigned int x = u << 16;
  float f; __builtin_memcpy(&f, &x, 4); return f;
}
__device__ __forceinline__ float bfhi(unsigned int u) {
  unsigned int x = u & 0xffff0000u;
  float f; __builtin_memcpy(&f, &x, 4); return f;
}
__device__ __forceinline__ unsigned short f2bf(float f) {
  unsigned int x; __builtin_memcpy(&x, &f, 4);
  x += 0x7FFFu + ((x >> 16) & 1u);
  return (unsigned short)(x >> 16);
}
__device__ __forceinline__ float sigmoidf_(float x) { return 1.0f / (1.0f + __expf(-x)); }

// ---------------------------------------------------------------------------
// WMMA fragment helpers (16x32 A / 32x16 B bf16, contiguous k = 16*(lane/16) + s in A and B, wave32)
// ---------------------------------------------------------------------------
__device__ __forceinline__ v16bf load_a_frag(const unsigned short* p) {
  // p points at this lane's K-base for the current 32-chunk (already + kA)
  v8bf lo = *reinterpret_cast<const v8bf*>(p);        // K: kA+0..7
  v8bf hi = *reinterpret_cast<const v8bf*>(p + 8);    // K: kA+8..15
  v16bf a;
#pragma unroll
  for (int i = 0; i < 8; ++i) { a[i] = lo[i]; a[i + 8] = hi[i]; }
  return a;
}

// ---------------------------------------------------------------------------
// Generic bf16 WMMA GEMM:  C[M,N] = A[M,K] * W[N,K]^T  (+ epilogue)
// One 16x16 tile per wave. K multiple of 32.
// mode 0: C=f32   mode 2: C = v + bias[n]   mode 3: Cb = bf16(tanh(v)) only
// ---------------------------------------------------------------------------
__global__ __launch_bounds__(128)
void k_gemm_bf16(const unsigned short* __restrict__ A,
                 const unsigned short* __restrict__ W,
                 float* __restrict__ C,
                 unsigned short* __restrict__ Cb,
                 const float* __restrict__ bias,
                 int M, int N, int K, int mode) {
  const int lane = threadIdx.x & 31;
  const int wave = threadIdx.x >> 5;
  const int nTilesN = N >> 4;
  const int tile = blockIdx.x * (blockDim.x >> 5) + wave;
  if (tile >= (M >> 4) * nTilesN) return;          // wave-uniform
  const int mt = tile / nTilesN;
  const int nt = tile % nTilesN;

  const int rowA = (mt << 4) + (lane & 15);
  const int kA   = (lane >> 4) << 4;               // 0 or 16
  const int colB = (nt << 4) + (lane & 15);
  const int kB   = (lane >> 4) << 4;               // 0 or 16

  const unsigned short* ap = A + (size_t)rowA * K + kA;
  const unsigned short* wp = W + (size_t)colB * K + kB;

  v8f acc = {};
  for (int k0 = 0; k0 < K; k0 += 32) {
    v16bf a = load_a_frag(ap + k0);
    v16bf b = *reinterpret_cast<const v16bf*>(wp + k0);
    acc = __builtin_amdgcn_wmma_f32_16x16x32_bf16(false, a, false, b,
                                                  (short)0, acc, false, false);
  }

  const int col   = (nt << 4) + (lane & 15);
  const int rbase = (mt << 4) + ((lane >> 4) << 3);
#pragma unroll
  for (int r = 0; r < 8; ++r) {
    float v = acc[r];
    size_t o = (size_t)(rbase + r) * N + col;
    if (mode == 3)      { Cb[o] = f2bf(tanhf(v)); }
    else if (mode == 2) { C[o] = v + bias[col]; }
    else                { C[o] = v; }
  }
}

// ---------------------------------------------------------------------------
// Fused LSTM-gates GEMM: gates[32,2048] = [xt | ctx_out | h] @ Wg^T
// A operand gathered on the fly (each K=32 chunk lies in one segment).
// ---------------------------------------------------------------------------
__global__ __launch_bounds__(128)
void k_gates_gemm(const unsigned short* __restrict__ xs,
                  const unsigned short* __restrict__ outB,
                  const unsigned short* __restrict__ hB,
                  const unsigned short* __restrict__ Wg,
                  float* __restrict__ gates, int t) {
  const int lane = threadIdx.x & 31;
  const int wave = threadIdx.x >> 5;
  const int tile = blockIdx.x * 4 + wave;          // 256 tiles: 2 x 128
  const int mt = tile >> 7;
  const int nt = tile & 127;

  const int b    = (mt << 4) + (lane & 15);        // batch row
  const int kA   = (lane >> 4) << 4;
  const int colB = (nt << 4) + (lane & 15);
  const int kB   = (lane >> 4) << 4;

  const unsigned short* xrow = xs + ((size_t)b * TT + t) * EE;
  const unsigned short* orow = outB + (size_t)b * HH;
  const unsigned short* hrow = hB + (size_t)b * HH;
  const unsigned short* wp   = Wg + (size_t)colB * KCAT + kB;

  v8f acc = {};
  for (int k0 = 0; k0 < KCAT; k0 += 32) {          // uniform segment select
    const unsigned short* src =
        (k0 < EE)      ? (xrow + k0) :
        (k0 < EE + HH) ? (orow + (k0 - EE)) :
                         (hrow + (k0 - EE - HH));
    v16bf a = load_a_frag(src + kA);
    v16bf bfrag = *reinterpret_cast<const v16bf*>(wp + k0);
    acc = __builtin_amdgcn_wmma_f32_16x16x32_bf16(false, a, false, bfrag,
                                                  (short)0, acc, false, false);
  }

  const int col   = (nt << 4) + (lane & 15);
  const int rbase = (mt << 4) + ((lane >> 4) << 3);
#pragma unroll
  for (int r = 0; r < 8; ++r)
    gates[(size_t)(rbase + r) * G4H + col] = acc[r];
}

// ---------------------------------------------------------------------------
// One-time prep kernels
// ---------------------------------------------------------------------------
__global__ void k_cat_weights(const float* __restrict__ w_ih,
                              const float* __restrict__ w_hh,
                              unsigned short* __restrict__ Wg) {
  int idx = blockIdx.x * 256 + threadIdx.x;
  if (idx >= G4H * KCAT) return;
  int n = idx / KCAT, k = idx % KCAT;
  float v = (k < EE + HH) ? w_ih[(size_t)n * (EE + HH) + k]
                          : w_hh[(size_t)n * HH + (k - (EE + HH))];
  Wg[idx] = f2bf(v);
}

__global__ void k_f32_to_bf16(const float* __restrict__ s,
                              unsigned short* __restrict__ d, int n) {
  int idx = blockIdx.x * 256 + threadIdx.x;
  if (idx < n) d[idx] = f2bf(s[idx]);
}

__global__ void k_embed(const float* __restrict__ embed_w,
                        const int* __restrict__ seq,
                        unsigned short* __restrict__ xs) {
  int idx = blockIdx.x * 256 + threadIdx.x;
  if (idx >= BB * TT * EE) return;
  int bt = idx / EE, e = idx % EE;
  int tok = seq[bt];
  float v = embed_w[(size_t)tok * EE + e];
  xs[idx] = f2bf(v > 0.0f ? v : 0.0f);
}

__global__ void k_zero_f32(float* p, int n) {
  int i = blockIdx.x * 256 + threadIdx.x; if (i < n) p[i] = 0.0f;
}
__global__ void k_zero_u16(unsigned short* p, int n) {
  int i = blockIdx.x * 256 + threadIdx.x; if (i < n) p[i] = 0;
}

// ---------------------------------------------------------------------------
// Per-step pointwise / reduction kernels
// ---------------------------------------------------------------------------
__global__ __launch_bounds__(256)
void k_lstm(const float* __restrict__ gates,
            float* __restrict__ c, float* __restrict__ h,
            unsigned short* __restrict__ hB,
            unsigned short* __restrict__ ch_cat) {
  int idx = blockIdx.x * 256 + threadIdx.x;
  if (idx >= BB * HH) return;
  int b = idx >> 9, j = idx & (HH - 1);
  const float* g = gates + (size_t)b * G4H;
  float ig = sigmoidf_(g[j]);
  float fg = sigmoidf_(g[HH + j]);
  float gg = tanhf(g[2 * HH + j]);
  float og = sigmoidf_(g[3 * HH + j]);
  float cn = fg * c[idx] + ig * gg;
  c[idx] = cn;
  float hn = og * tanhf(cn);
  h[idx] = hn;
  unsigned short hb = f2bf(hn);
  hB[idx] = hb;
  ch_cat[(size_t)b * (2 * HH) + HH + j] = hb;   // second half of [ctx|h]
}

// attention: one workgroup per batch row. uint4-vectorized scores, packed ctx.
__global__ __launch_bounds__(256)
void k_attention(const unsigned short* __restrict__ cnn_b,
                 const float* __restrict__ mapped,
                 unsigned short* __restrict__ ch_cat) {
  __shared__ float s_map[HH];
  __shared__ float s_sc[LL];
  __shared__ float s_red[256];
  const int b = blockIdx.x, tid = threadIdx.x;

  for (int hh = tid; hh < HH; hh += 256) s_map[hh] = mapped[(size_t)b * HH + hh];
  __syncthreads();

  const unsigned short* base = cnn_b + (size_t)b * LL * HH;
  for (int l = tid; l < LL; l += 256) {
    const uint4* rp = reinterpret_cast<const uint4*>(base + (size_t)l * HH);
    float s = 0.0f;
#pragma unroll 4
    for (int q = 0; q < HH / 8; ++q) {           // 8 bf16 per uint4
      uint4 u = rp[q];
      const float* mp = s_map + q * 8;
      s = fmaf(bflo(u.x), mp[0], s); s = fmaf(bfhi(u.x), mp[1], s);
      s = fmaf(bflo(u.y), mp[2], s); s = fmaf(bfhi(u.y), mp[3], s);
      s = fmaf(bflo(u.z), mp[4], s); s = fmaf(bfhi(u.z), mp[5], s);
      s = fmaf(bflo(u.w), mp[6], s); s = fmaf(bfhi(u.w), mp[7], s);
    }
    s_sc[l] = s;
  }
  __syncthreads();

  float m = -1e30f;
  for (int l = tid; l < LL; l += 256) m = fmaxf(m, s_sc[l]);
  s_red[tid] = m; __syncthreads();
  for (int s = 128; s > 0; s >>= 1) {
    if (tid < s) s_red[tid] = fmaxf(s_red[tid], s_red[tid + s]);
    __syncthreads();
  }
  float gmax = s_red[0]; __syncthreads();

  float lsum = 0.0f;
  for (int l = tid; l < LL; l += 256) {
    float e = __expf(s_sc[l] - gmax);
    s_sc[l] = e; lsum += e;
  }
  s_red[tid] = lsum; __syncthreads();
  for (int s = 128; s > 0; s >>= 1) {
    if (tid < s) s_red[tid] += s_red[tid + s];
    __syncthreads();
  }
  float inv = 1.0f / s_red[0];

  // ctx: each thread owns 2 adjacent H positions -> one coalesced u32 per l
  {
    const int hh = tid * 2;                        // HH = 2*256
    const unsigned int* cp =
        reinterpret_cast<const unsigned int*>(base) + (hh >> 1);
    float a0 = 0.0f, a1 = 0.0f;
    for (int l = 0; l < LL; ++l) {
      unsigned int u = cp[(size_t)l * (HH / 2)];
      float w = s_sc[l];
      a0 = fmaf(bflo(u), w, a0);
      a1 = fmaf(bfhi(u), w, a1);
    }
    a0 *= inv; a1 *= inv;
    unsigned int packed = (unsigned int)f2bf(a0) | ((unsigned int)f2bf(a1) << 16);
    reinterpret_cast<unsigned int*>(ch_cat)[((size_t)b * (2 * HH) + hh) >> 1] = packed;
  }
}

__global__ __launch_bounds__(256)
void k_logsoftmax(const float* __restrict__ logits,
                  float* __restrict__ out, int t) {
  __shared__ float s_red[256];
  const int b = blockIdx.x, tid = threadIdx.x;
  const float* row = logits + (size_t)b * VV;

  float m = -1e30f;
  for (int v = tid; v < VV; v += 256) m = fmaxf(m, row[v]);
  s_red[tid] = m; __syncthreads();
  for (int s = 128; s > 0; s >>= 1) {
    if (tid < s) s_red[tid] = fmaxf(s_red[tid], s_red[tid + s]);
    __syncthreads();
  }
  float gmax = s_red[0]; __syncthreads();

  float lsum = 0.0f;
  for (int v = tid; v < VV; v += 256) lsum += __expf(row[v] - gmax);
  s_red[tid] = lsum; __syncthreads();
  for (int s = 128; s > 0; s >>= 1) {
    if (tid < s) s_red[tid] += s_red[tid + s];
    __syncthreads();
  }
  float lse = gmax + __logf(s_red[0]);

  float* o = out + ((size_t)b * TT + t) * VV;
  for (int v = tid; v < VV; v += 256) o[v] = row[v] - lse;
}

// ---------------------------------------------------------------------------
extern "C" void kernel_launch(void* const* d_in, const int* in_sizes, int n_in,
                              void* d_out, int out_size, void* d_ws, size_t ws_size,
                              hipStream_t stream) {
  const float* cnn_feats = (const float*)d_in[0];
  const int*   seq       = (const int*)d_in[1];
  const float* embed_w   = (const float*)d_in[2];
  const float* w_ih      = (const float*)d_in[3];
  const float* w_hh      = (const float*)d_in[4];
  const float* w_hm      = (const float*)d_in[5];
  const float* w_out     = (const float*)d_in[6];
  const float* w_logit   = (const float*)d_in[7];
  const float* b_logit   = (const float*)d_in[8];
  float* outp = (float*)d_out;

  char* ws = (char*)d_ws;
  size_t off = 0;
  auto alloc = [&](size_t bytes) -> char* {
    char* p = ws + off;
    off = (off + bytes + 255) & ~(size_t)255;
    return p;
  };

  unsigned short* xs     = (unsigned short*)alloc((size_t)BB * TT * EE * 2);
  unsigned short* Wg     = (unsigned short*)alloc((size_t)G4H * KCAT * 2);
  unsigned short* whm    = (unsigned short*)alloc((size_t)HH * HH * 2);
  unsigned short* wout   = (unsigned short*)alloc((size_t)HH * 2 * HH * 2);
  unsigned short* wlog   = (unsigned short*)alloc((size_t)VV * HH * 2);
  unsigned short* cnnb   = (unsigned short*)alloc((size_t)BB * LL * HH * 2);
  float*          hF     = (float*)alloc((size_t)BB * HH * 4);
  float*          cF     = (float*)alloc((size_t)BB * HH * 4);
  unsigned short* hB     = (unsigned short*)alloc((size_t)BB * HH * 2);
  unsigned short* outB   = (unsigned short*)alloc((size_t)BB * HH * 2);
  unsigned short* chcat  = (unsigned short*)alloc((size_t)BB * 2 * HH * 2);
  float*          gates  = (float*)alloc((size_t)BB * G4H * 4);
  float*          mapped = (float*)alloc((size_t)BB * HH * 4);
  float*          logits = (float*)alloc((size_t)BB * VV * 4);

  auto blk = [](int n) { return (n + 255) / 256; };

  // one-time prep: bf16 conversions, weight concat, embedding+relu, state init
  k_cat_weights<<<blk(G4H * KCAT), 256, 0, stream>>>(w_ih, w_hh, Wg);
  k_f32_to_bf16<<<blk(HH * HH), 256, 0, stream>>>(w_hm, whm, HH * HH);
  k_f32_to_bf16<<<blk(HH * 2 * HH), 256, 0, stream>>>(w_out, wout, HH * 2 * HH);
  k_f32_to_bf16<<<blk(VV * HH), 256, 0, stream>>>(w_logit, wlog, VV * HH);
  k_f32_to_bf16<<<blk(BB * LL * HH), 256, 0, stream>>>(cnn_feats, cnnb, BB * LL * HH);
  k_embed<<<blk(BB * TT * EE), 256, 0, stream>>>(embed_w, seq, xs);
  k_zero_f32<<<blk(BB * HH), 256, 0, stream>>>(hF, BB * HH);
  k_zero_f32<<<blk(BB * HH), 256, 0, stream>>>(cF, BB * HH);
  k_zero_u16<<<blk(BB * HH), 256, 0, stream>>>(hB, BB * HH);
  k_zero_u16<<<blk(BB * HH), 256, 0, stream>>>(outB, BB * HH);
  k_zero_u16<<<blk(BB * 2 * HH), 256, 0, stream>>>(chcat, BB * 2 * HH);

  auto gemm = [&](const unsigned short* A, const unsigned short* W, float* C,
                  unsigned short* Cb, const float* bias,
                  int M, int N, int K, int mode) {
    int tiles = (M / 16) * (N / 16);
    int blocks = (tiles + 3) / 4;                 // 4 waves (128 threads) / block
    k_gemm_bf16<<<blocks, 128, 0, stream>>>(A, W, C, Cb, bias, M, N, K, mode);
  };

  for (int t = 0; t < TT; ++t) {
    k_gates_gemm<<<64, 128, 0, stream>>>(xs, outB, hB, Wg, gates, t);   // fused gather+GEMM
    k_lstm<<<blk(BB * HH), 256, 0, stream>>>(gates, cF, hF, hB, chcat);
    gemm(hB, whm, mapped, nullptr, nullptr, BB, HH, HH, 0);             // h @ w_hm^T
    k_attention<<<BB, 256, 0, stream>>>(cnnb, mapped, chcat);           // softmax attn
    gemm(chcat, wout, nullptr, outB, nullptr, BB, HH, 2 * HH, 3);       // tanh -> bf16
    gemm(outB, wlog, logits, nullptr, b_logit, BB, VV, HH, 2);          // logits + bias
    k_logsoftmax<<<BB, 256, 0, stream>>>(logits, outp, t);
  }
}
